// GNNDecoder_14482629722146
// MI455X (gfx1250) — matrix operand order repaired and checked by reading, not compiled, not run on hardware
//
#include <hip/hip_runtime.h>

// ---------------- problem constants (from reference) ----------------
#define NNODES 30000
#define NEDGES 960000
#define INC    256
#define HIDC   128
#define OUTC   64
#define NRNA   6000
#define NDRUG  24000   // NNODES - NRNA

// ---------------- WMMA vector types ----------------
typedef __attribute__((ext_vector_type(16))) __bf16          v16bf;
typedef __attribute__((ext_vector_type(8)))  float           v8f;
typedef __attribute__((ext_vector_type(8)))  unsigned short  v8u16;
typedef __attribute__((ext_vector_type(16))) unsigned short  v16u16;

union BF16Vec {
  v16bf v;
  unsigned short s[16];
  v16u16 u;
  struct { v8u16 lo; v8u16 hi; } h;
};

__device__ __forceinline__ unsigned short f2bf(float f) {
  union { float f; unsigned int u; } c;
  c.f = f;
  unsigned int u = c.u;
  u += 0x7FFFu + ((u >> 16) & 1u);   // round-to-nearest-even
  return (unsigned short)(u >> 16);
}

// ---------------- zero workspace ----------------
__global__ void k_zero4(float4* __restrict__ p, int n4) {
  int i = blockIdx.x * blockDim.x + threadIdx.x;
  if (i < n4) p[i] = make_float4(0.f, 0.f, 0.f, 0.f);
}

// ---------------- edge scatter-add (segment_sum) ----------------
// One thread per (edge, 4-channel group). Accumulators live in L2 (~31MB).
template<int C4, int C, bool ADDCNT>
__global__ void k_scatter(const float* __restrict__ xin,
                          const long long* __restrict__ eidx,
                          float* __restrict__ agg,
                          float* __restrict__ cnt) {
  int tid = blockIdx.x * blockDim.x + threadIdx.x;   // grid sized exactly
  int e = tid / C4;          // C4 is power of two -> shift
  int c = tid - e * C4;
  int src = (int)eidx[e];
  int dst = (int)eidx[NEDGES + e];
  float4 v = ((const float4*)xin)[src * C4 + c];
  float* d = agg + (size_t)dst * C + c * 4;
  atomicAdd(d + 0, v.x);
  atomicAdd(d + 1, v.y);
  atomicAdd(d + 2, v.z);
  atomicAdd(d + 3, v.w);
  if (ADDCNT && c == 0) atomicAdd(cnt + dst, 1.0f);
}

// ---------------- fused SAGE GEMM:  out = act( (agg/deg)@Wl + b + xin@Wr ) ----
// One wave32 computes one 16x16 f32 tile via v_wmma_f32_16x16x32_bf16.
// BF16OUT stores the result as bf16 (row-major) for the decoder's fast path.
template<int K1, int N, int NTN, bool RELU, bool BF16OUT>
__global__ void k_sage_gemm(const float* __restrict__ agg,
                            const float* __restrict__ cnt,
                            const float* __restrict__ xin,
                            const float* __restrict__ Wl,
                            const float* __restrict__ Wr,
                            const float* __restrict__ bias,
                            void* __restrict__ outv) {
  int wave = (blockIdx.x * blockDim.x + threadIdx.x) >> 5;
  int lane = threadIdx.x & 31;
  int tm = wave / NTN, tn = wave - tm * NTN;
  int l16 = lane & 15;
  int hi  = lane >> 4;
  int rowA = tm * 16 + l16;          // A row held by this lane
  int colB = tn * 16 + l16;          // B/D column held by this lane
  float deg = cnt[rowA];
  float scale = 1.0f / fmaxf(deg, 1.0f);

  v8f acc = {};
  #pragma unroll
  for (int term = 0; term < 2; ++term) {
    const float* A = term ? xin : agg;
    const float* B = term ? Wr  : Wl;
    float s = term ? 1.0f : scale;
    for (int k0 = 0; k0 < K1; k0 += 32) {
      BF16Vec a, b;
      #pragma unroll
      for (int i = 0; i < 16; ++i) {
        int ka = k0 + ((i >> 3) << 4) + (i & 7) + (hi << 3);   // A 16x32 bf16 layout
        a.s[i] = f2bf(A[(size_t)rowA * K1 + ka] * s);
        int kb = k0 + ((i >> 3) << 4) + (i & 7) + (hi << 3);   // B 32x16 bf16 layout
        b.s[i] = f2bf(B[(size_t)kb * N + colB]);
      }
      acc = __builtin_amdgcn_wmma_f32_16x16x32_bf16(
          false, a.v, false, b.v, (short)0, acc, false, false);
    }
  }
  float bv = bias[colB];
  #pragma unroll
  for (int r = 0; r < 8; ++r) {
    float v = acc[r] + bv;
    if (RELU) v = fmaxf(v, 0.0f);
    size_t idx = (size_t)(tm * 16 + (hi << 3) + r) * N + colB;  // C/D layout
    if (BF16OUT) ((unsigned short*)outv)[idx] = f2bf(v);
    else         ((float*)outv)[idx] = v;
  }
}

// ---------------- decoder: out[i][j] = sum_k h2[i][k] * h2[NRNA+j][k] -------
// h2 is pre-converted bf16 row-major (rows of 64 = 128B). Each wave computes a
// 16x64 strip (4 N-tiles) so A is loaded once per K-step and reused across 4
// WMMAs; operand loads are pure b128 vector loads (no conversion VALU).
// Non-temporal stores: 576MB output stream must not evict L2-resident h2.
__global__ void k_decoder(const unsigned short* __restrict__ h2b,
                          float* __restrict__ out) {
  const int NTG = NDRUG / 64;   // 375 groups of 4 N-tiles
  int wave = (blockIdx.x * blockDim.x + threadIdx.x) >> 5;
  int lane = threadIdx.x & 31;
  int tm = wave / NTG, tg = wave - tm * NTG;
  int l16 = lane & 15;
  int hi  = lane >> 4;
  const unsigned short* pa = h2b + (size_t)(tm * 16 + l16) * OUTC;

  v8f acc[4] = {{}, {}, {}, {}};
  #pragma unroll
  for (int k0 = 0; k0 < OUTC; k0 += 32) {
    BF16Vec a;
    a.h.lo = *(const v8u16*)(pa + k0 + (hi << 3));        // A elems 0..7
    a.h.hi = *(const v8u16*)(pa + k0 + 16 + (hi << 3));   // A elems 8..15
    #pragma unroll
    for (int t = 0; t < 4; ++t) {
      int rowB = NRNA + (tg * 4 + t) * 16 + l16;          // drug node
      BF16Vec b;
      b.h.lo = *(const v8u16*)(h2b + (size_t)rowB * OUTC + k0 + (hi << 3));
      b.h.hi = *(const v8u16*)(h2b + (size_t)rowB * OUTC + k0 + 16 + (hi << 3));
      acc[t] = __builtin_amdgcn_wmma_f32_16x16x32_bf16(
          false, a.v, false, b.v, (short)0, acc[t], false, false);
    }
  }
  #pragma unroll
  for (int t = 0; t < 4; ++t) {
    int col = (tg * 4 + t) * 16 + l16;
    #pragma unroll
    for (int r = 0; r < 8; ++r) {
      float* p = out + (size_t)(tm * 16 + (hi << 3) + r) * NDRUG + col;
      __builtin_nontemporal_store(acc[t][r], p);
    }
  }
}

// ---------------- host launch ----------------
extern "C" void kernel_launch(void* const* d_in, const int* in_sizes, int n_in,
                              void* d_out, int out_size, void* d_ws, size_t ws_size,
                              hipStream_t stream) {
  const float*     x    = (const float*)d_in[0];
  const long long* eidx = (const long long*)d_in[1];   // int64 edge_index [2, NEDGES]
  // d_in[2] = num_RNA (compile-time constant 6000, unused at runtime)
  const float* W1l = (const float*)d_in[3];
  const float* W1r = (const float*)d_in[4];
  const float* b1  = (const float*)d_in[5];
  const float* W2l = (const float*)d_in[6];
  const float* W2r = (const float*)d_in[7];
  const float* b2  = (const float*)d_in[8];
  float* out = (float*)d_out;

  // workspace layout (floats; every offset is a multiple of 4 floats = 16B)
  float* ws   = (float*)d_ws;
  float* agg1 = ws;                                   // 30000*256 = 7,680,000
  float* cnt  = agg1 + (size_t)NNODES * INC;          // 30,000
  float* h1   = cnt  + NNODES;                        // 30000*128 = 3,840,000
  float* agg2 = h1   + (size_t)NNODES * HIDC;         // 3,840,000
  unsigned short* h2b = (unsigned short*)(agg2 + (size_t)NNODES * HIDC); // 30000*64 bf16
  // total: ~15.39M floats + 3.84MB bf16 (~65.4 MB)

  // 1) zero agg1 + cnt + h1 + agg2 (h1 cleared harmlessly; fully overwritten)
  {
    int n4 = (7680000 + 30000 + 3840000 + 3840000) / 4;   // 3,847,500
    k_zero4<<<(n4 + 255) / 256, 256, 0, stream>>>((float4*)ws, n4);
  }
  // 2) layer-1 scatter: 960000 edges * 64 float4 groups = 61,440,000 threads
  k_scatter<INC / 4, INC, true>
      <<<NEDGES * (INC / 4) / 256, 256, 0, stream>>>(x, eidx, agg1, cnt);
  // 3) fused GEMM 1: 1875*8 = 15000 waves, 8 waves/block
  k_sage_gemm<INC, HIDC, HIDC / 16, true, false>
      <<<15000 / 8, 256, 0, stream>>>(agg1, cnt, x, W1l, W1r, b1, h1);
  // 4) layer-2 scatter: 960000 * 32 = 30,720,000 threads (degrees unchanged)
  k_scatter<HIDC / 4, HIDC, false>
      <<<NEDGES * (HIDC / 4) / 256, 256, 0, stream>>>(h1, eidx, agg2, cnt);
  // 5) fused GEMM 2 -> bf16 h2: 1875*4 = 7500 waves, 4 waves/block
  k_sage_gemm<HIDC, OUTC, OUTC / 16, false, true>
      <<<7500 / 4, 128, 0, stream>>>(agg2, cnt, h1, W2l, W2r, b2, (void*)h2b);
  // 6) decoder: 375 * 375 = 140,625 waves, 1 wave/block (exact grid)
  k_decoder<<<140625, 32, 0, stream>>>(h2b, out);
}
